// Decoder_AdptiveVisualCenter_codebook_35536559407361
// MI455X (gfx1250) — compile-verified, matrix operand order checked by reading, not run on hardware
//
#include <hip/hip_runtime.h>
#include <hip/hip_bf16.h>

// ---------------------------------------------------------------------------
// Problem constants (from reference): B=16, C=256, H=W=64, CR=32, K=16
// ---------------------------------------------------------------------------
#define BB    16
#define CC    256
#define CRr   32
#define NPIX  4096     // 64*64
#define KCB   16
#define HH    64
#define WW    64

typedef __attribute__((ext_vector_type(16))) _Float16 v16h;
typedef __attribute__((ext_vector_type(8)))  _Float16 v8h;
typedef __attribute__((ext_vector_type(8)))  float    v8f;
typedef __attribute__((ext_vector_type(4)))  float    v4f;
typedef __attribute__((ext_vector_type(4)))  int      i4;

// Optional CDNA5 async global->LDS path (ASYNCcnt), with safe fallback.
#if defined(__HIP_DEVICE_COMPILE__) &&                                   \
    __has_builtin(__builtin_amdgcn_global_load_async_to_lds_b128) &&     \
    __has_builtin(__builtin_amdgcn_s_wait_asynccnt)
#define USE_ASYNC_LDS 1
#else
#define USE_ASYNC_LDS 0
#endif

#if USE_ASYNC_LDS
typedef __attribute__((address_space(1))) i4 gi4;   // global int4
typedef __attribute__((address_space(3))) i4 li4;   // LDS int4
#endif

// Wave-local LDS ordering: LDS ops from one wave execute in order (ISA §7.3),
// and the compiler inserts s_wait_dscnt for VGPR RAW hazards.  We only need a
// compiler reordering fence -- no workgroup barrier, so the 8 waves per block
// are free-running.
__device__ __forceinline__ void lds_fence() { asm volatile("" ::: "memory"); }

// Stage 32 bytes (16 f16) from global into LDS.
__device__ __forceinline__ void stage_a32B(const _Float16* __restrict__ gsrc,
                                           _Float16* ldst) {
#if USE_ASYNC_LDS
  gi4* g4 = (gi4*)(const void*)gsrc;
  li4* l4 = (li4*)(void*)ldst;
  __builtin_amdgcn_global_load_async_to_lds_b128(g4,     l4,     0, 0);
  __builtin_amdgcn_global_load_async_to_lds_b128(g4 + 1, l4 + 1, 0, 0);
#else
  const uint4* s4 = (const uint4*)gsrc;
  *(uint4*)(ldst)     = s4[0];
  *(uint4*)(ldst + 8) = s4[1];
#endif
}

__device__ __forceinline__ void stage_wait() {
#if USE_ASYNC_LDS
  __builtin_amdgcn_s_wait_asynccnt(0);
#endif
}

// ---------------------------------------------------------------------------
// WMMA helpers.
// A fragment (16x32 f16): lane L<16 holds row M=L, K = {kg..kg+7, 16+kg..16+kg+7}
// with kg = (L>=16)*8.  B fragment mirrors with N instead of M, so we stage B
// transposed in LDS ([col][32 k]) and both fragments become two b128 LDS reads.
// C/D (16x16 f32): VGPR r of lane L -> (M = r + (L>=16)*8, N = L&15).
// ---------------------------------------------------------------------------
__device__ __forceinline__ void wmma_gather(const _Float16* lA, const _Float16* lBt,
                                            v16h& a, v16h& b) {
  const int lane = threadIdx.x & 31;
  const int rm = lane & 15;
  const int kg = (lane >> 4) << 3;
  v8h a0 = *(const v8h*)(lA  + rm * 32 + kg);
  v8h a1 = *(const v8h*)(lA  + rm * 32 + 16 + kg);
  v8h b0 = *(const v8h*)(lBt + rm * 32 + kg);
  v8h b1 = *(const v8h*)(lBt + rm * 32 + 16 + kg);
#pragma unroll
  for (int j = 0; j < 8; ++j) {
    a[j] = a0[j]; a[j + 8] = a1[j];
    b[j] = b0[j]; b[j + 8] = b1[j];
  }
}

__device__ __forceinline__ v8f wmma_step(v16h a, v16h b, v8f c) {
  return __builtin_amdgcn_wmma_f32_16x16x32_f16(false, a, false, b, (short)0, c,
                                                false, false);
}

// ---------------------------------------------------------------------------
// fp32 -> fp16 conversion (grid-stride)
// ---------------------------------------------------------------------------
__global__ void cvt_f32_f16_kernel(const float* __restrict__ src,
                                   _Float16* __restrict__ dst, int n) {
  for (int i = blockIdx.x * blockDim.x + threadIdx.x; i < n;
       i += gridDim.x * blockDim.x)
    dst[i] = (_Float16)src[i];
}

// Pack OIHW 3x3 weights -> [tap][Cout][Cin] f16 so each tap's A-tile is dense.
__global__ void pack_w3x3_kernel(const float* __restrict__ src,
                                 _Float16* __restrict__ dst, int Cout, int Cin) {
  int total = 9 * Cout * Cin;
  for (int i = blockIdx.x * blockDim.x + threadIdx.x; i < total;
       i += gridDim.x * blockDim.x) {
    int t = i / (Cout * Cin);
    int rem = i - t * Cout * Cin;
    int o = rem / Cin, ci = rem - o * Cin;
    dst[i] = (_Float16)src[(size_t)(o * Cin + ci) * 9 + t];
  }
}

// ---------------------------------------------------------------------------
// 1x1 conv as GEMM:  out[b][Cout][N] = W[Cout,Cin] * in[b][Cin][N] + bias
// one wave = one 16x16 output tile.  grid: (N/16/8, Cout/16, B), block 256.
// ---------------------------------------------------------------------------
__global__ void conv1x1_wmma_kernel(const _Float16* __restrict__ in,
                                    const _Float16* __restrict__ w,
                                    const float* __restrict__ bias,
                                    _Float16* __restrict__ out,
                                    int Cin, int Cout) {
  __shared__ alignas(16) _Float16 sA[8][512];
  __shared__ alignas(16) _Float16 sB[8][512];
  const int wave = threadIdx.x >> 5;
  const int lane = threadIdx.x & 31;
  const int n0 = (blockIdx.x * 8 + wave) * 16;
  const int m0 = blockIdx.y * 16;
  const int b  = blockIdx.z;
  const int rm = lane & 15, hf = lane >> 4;
  const _Float16* inb = in + (size_t)b * Cin * NPIX;
  v8f acc = {};
  for (int kc = 0; kc < Cin; kc += 32) {
    // A: weights, 16 rows x 32 k, row-major -> async copy straight to LDS.
    stage_a32B(w + (size_t)(m0 + rm) * Cin + kc + hf * 16,
               &sA[wave][rm * 32 + hf * 16]);
    // B: activations, 32 k x 16 px, stored transposed [px][32].
    const _Float16* bsrc = inb + (size_t)(kc + lane) * NPIX + n0;
    uint4 t0 = *(const uint4*)bsrc;
    uint4 t1 = *(const uint4*)(bsrc + 8);
    _Float16 tmp[16];
    *(uint4*)(tmp)     = t0;
    *(uint4*)(tmp + 8) = t1;
#pragma unroll
    for (int j = 0; j < 16; ++j) sB[wave][j * 32 + lane] = tmp[j];
    stage_wait();
    lds_fence();
    v16h av, bv;
    wmma_gather(sA[wave], sB[wave], av, bv);
    acc = wmma_step(av, bv, acc);
    lds_fence();
  }
  const int col = lane & 15, hi = lane >> 4;
#pragma unroll
  for (int r = 0; r < 8; ++r) {
    int m = m0 + r + hi * 8;
    float v = acc[r] + bias[m];
    out[(size_t)b * Cout * NPIX + (size_t)m * NPIX + n0 + col] = (_Float16)v;
  }
}

// ---------------------------------------------------------------------------
// 3x3 conv (pad 1) as implicit GEMM over 9 taps, optional fused BN+ReLU.
// wp layout: [tap][Cout][Cin].  N-tile = 16 px within one image row.
// ---------------------------------------------------------------------------
__global__ void conv3x3_wmma_kernel(const _Float16* __restrict__ in,
                                    const _Float16* __restrict__ wp,
                                    const float* __restrict__ bias,
                                    const float* __restrict__ gamma,
                                    const float* __restrict__ beta,
                                    const float* __restrict__ mean,
                                    const float* __restrict__ var,
                                    _Float16* __restrict__ out,
                                    int Cin, int Cout, int do_bnrelu) {
  __shared__ alignas(16) _Float16 sA[8][512];
  __shared__ alignas(16) _Float16 sB[8][512];
  const int wave = threadIdx.x >> 5;
  const int lane = threadIdx.x & 31;
  const int n0 = (blockIdx.x * 8 + wave) * 16;
  const int m0 = blockIdx.y * 16;
  const int b  = blockIdx.z;
  const int h0 = n0 >> 6;
  const int w0 = n0 & 63;
  const int rm = lane & 15, hf = lane >> 4;
  const _Float16* inb = in + (size_t)b * Cin * NPIX;
  v8f acc = {};
#pragma unroll
  for (int t = 0; t < 9; ++t) {
    const int dy = t / 3 - 1, dx = t % 3 - 1;
    const int hs = h0 + dy;
    const bool hok = (unsigned)hs < (unsigned)HH;
    for (int kc = 0; kc < Cin; kc += 32) {
      stage_a32B(wp + ((size_t)t * Cout + (m0 + rm)) * Cin + kc + hf * 16,
                 &sA[wave][rm * 32 + hf * 16]);
      const _Float16* bsrc =
          inb + (size_t)(kc + lane) * NPIX + hs * WW + w0 + dx;
      __builtin_prefetch(bsrc + 32 * NPIX, 0, 1);  // next k-step
      if (dx == 0 && hok) {
        // interior row, 32B-aligned contiguous -> 2 x b128
        uint4 t0 = *(const uint4*)bsrc;
        uint4 t1 = *(const uint4*)(bsrc + 8);
        _Float16 tmp[16];
        *(uint4*)(tmp)     = t0;
        *(uint4*)(tmp + 8) = t1;
#pragma unroll
        for (int j = 0; j < 16; ++j) sB[wave][j * 32 + lane] = tmp[j];
      } else {
#pragma unroll
        for (int j = 0; j < 16; ++j) {
          int ws = w0 + dx + j;
          _Float16 v = (_Float16)0.f;
          if (hok && (unsigned)ws < (unsigned)WW) v = bsrc[j];
          sB[wave][j * 32 + lane] = v;
        }
      }
      stage_wait();
      lds_fence();
      v16h av, bv;
      wmma_gather(sA[wave], sB[wave], av, bv);
      acc = wmma_step(av, bv, acc);
      lds_fence();
    }
  }
  const int col = lane & 15, hi = lane >> 4;
#pragma unroll
  for (int r = 0; r < 8; ++r) {
    int m = m0 + r + hi * 8;
    float v = acc[r] + bias[m];
    if (do_bnrelu) {
      float inv = rsqrtf(var[m] + 1e-5f);
      v = (v - mean[m]) * (gamma[m] * inv) + beta[m];
      v = fmaxf(v, 0.f);
    }
    out[(size_t)b * Cout * NPIX + (size_t)m * NPIX + n0 + col] = (_Float16)v;
  }
}

// ---------------------------------------------------------------------------
// ||xf||^2 per pixel and ||codebook||^2 per codeword
// ---------------------------------------------------------------------------
__global__ void xnorm_kernel(const _Float16* __restrict__ xch,
                             float* __restrict__ xn) {
  int b = blockIdx.y;
  int n = blockIdx.x * blockDim.x + threadIdx.x;
  const _Float16* p = xch + (size_t)b * CC * NPIX + n;
  float s = 0.f;
  for (int c = 0; c < CC; ++c) {
    float v = (float)p[(size_t)c * NPIX];
    s += v * v;
  }
  xn[(size_t)b * NPIX + n] = s;
}

__global__ void cnorm_kernel(const float* __restrict__ cb,
                             float* __restrict__ cn) {
  int k = threadIdx.x;
  if (k < KCB) {
    float s = 0.f;
    for (int c = 0; c < CC; ++c) {
      float v = cb[k * CC + c];
      s += v * v;
    }
    cn[k] = s;
  }
}

// ---------------------------------------------------------------------------
// Distance GEMM (16 px x 16 codewords tile) + softmax over K via wave32
// shuffles.  d2 = |x|^2 + |c|^2 - 2 x.c ; w = softmax(-s_k * sqrt(d2)).
// ---------------------------------------------------------------------------
__global__ void dist_softmax_kernel(const _Float16* __restrict__ xch,
                                    const float* __restrict__ cb,
                                    const float* __restrict__ xn,
                                    const float* __restrict__ cn,
                                    const float* __restrict__ sm,
                                    float* __restrict__ wt) {
  __shared__ alignas(16) _Float16 sA[8][512];
  __shared__ alignas(16) _Float16 sB[8][512];
  const int wave = threadIdx.x >> 5, lane = threadIdx.x & 31;
  const int n0 = (blockIdx.x * 8 + wave) * 16;
  const int b = blockIdx.z;
  const int rm = lane & 15, hf = lane >> 4;
  const _Float16* xb = xch + (size_t)b * CC * NPIX;
  v8f acc = {};
  for (int kc = 0; kc < CC; kc += 32) {
#pragma unroll
    for (int j = 0; j < 16; ++j) {
      int c = hf * 16 + j;
      sA[wave][rm * 32 + c] = xb[(size_t)(kc + c) * NPIX + n0 + rm];
      sB[wave][j * 32 + lane] = (_Float16)cb[j * CC + kc + lane];
    }
    lds_fence();
    v16h av, bv;
    wmma_gather(sA[wave], sB[wave], av, bv);
    acc = wmma_step(av, bv, acc);
    lds_fence();
  }
  const int col = lane & 15, hi = lane >> 4;
  float smv = sm[col], cnv = cn[col];
#pragma unroll
  for (int r = 0; r < 8; ++r) {
    int n = n0 + r + hi * 8;
    float d2 = xn[(size_t)b * NPIX + n] + cnv - 2.f * acc[r];
    float lg = -smv * sqrtf(fmaxf(d2, 1e-12f));
    float mx = lg;
#pragma unroll
    for (int msk = 1; msk < 16; msk <<= 1)
      mx = fmaxf(mx, __shfl_xor(mx, msk, 16));
    float ex = __expf(lg - mx);
    float s = ex;
#pragma unroll
    for (int msk = 1; msk < 16; msk <<= 1) s += __shfl_xor(s, msk, 16);
    wt[((size_t)b * NPIX + n) * KCB + col] = ex / s;
  }
}

// ---------------------------------------------------------------------------
// e_k[b][16][256] = W^T[16,4096] * Xf[4096,256]  (K-dim = pixels)
// ---------------------------------------------------------------------------
__global__ void ek_wmma_kernel(const float* __restrict__ wt,
                               const _Float16* __restrict__ xch,
                               float* __restrict__ ek) {
  __shared__ alignas(16) _Float16 sA[8][512];
  __shared__ alignas(16) _Float16 sB[8][512];
  const int wave = threadIdx.x >> 5, lane = threadIdx.x & 31;
  const int c0 = (blockIdx.x * 8 + wave) * 16;
  const int b = blockIdx.z;
  const int rm = lane & 15, hf = lane >> 4;
  const _Float16* xb = xch + (size_t)b * CC * NPIX;
  const float* wb = wt + (size_t)b * NPIX * KCB;
  v8f acc = {};
  for (int p0 = 0; p0 < NPIX; p0 += 32) {
#pragma unroll
    for (int j = 0; j < 16; ++j) {
      int p = hf * 16 + j;
      sA[wave][rm * 32 + p] = (_Float16)wb[(size_t)(p0 + p) * KCB + rm];
      sB[wave][j * 32 + lane] = xb[(size_t)(c0 + j) * NPIX + p0 + lane];
    }
    lds_fence();
    v16h av, bv;
    wmma_gather(sA[wave], sB[wave], av, bv);
    acc = wmma_step(av, bv, acc);
    lds_fence();
  }
  const int col = lane & 15, hi = lane >> 4;
#pragma unroll
  for (int r = 0; r < 8; ++r) {
    int k = r + hi * 8;
    ek[(size_t)b * KCB * CC + (size_t)k * CC + c0 + col] = acc[r];
  }
}

// ---------------------------------------------------------------------------
// mean over K, FC (256x256 matvec per batch), sigmoid
// ---------------------------------------------------------------------------
__global__ void fc_sigmoid_kernel(const float* __restrict__ ek,
                                  const float* __restrict__ fcw,
                                  const float* __restrict__ fcb,
                                  float* __restrict__ ev) {
  __shared__ float m[CC];
  const int b = blockIdx.x, t = threadIdx.x;
  float s = 0.f;
#pragma unroll
  for (int k = 0; k < KCB; ++k) s += ek[(size_t)b * KCB * CC + k * CC + t];
  m[t] = s * (1.f / (float)KCB);
  __syncthreads();  // real cross-wave dependency here
  float acc = fcb[t];
  for (int c = 0; c < CC; ++c) acc += m[c] * fcw[t * CC + c];
  ev[b * CC + t] = 1.f / (1.f + __expf(-acc));
}

// out = x * (1 + e[b,c]) , vectorized, non-temporal store
__global__ void gate_kernel(const float* __restrict__ x,
                            const float* __restrict__ ev,
                            float* __restrict__ out) {
  size_t i = (size_t)blockIdx.x * blockDim.x + threadIdx.x;
  size_t e = i * 4;
  int c = (int)((e >> 12) & 255);   // NPIX = 4096
  int b = (int)(e >> 20);           // C*NPIX = 1<<20
  float g = 1.f + ev[b * CC + c];
  v4f xv = *(const v4f*)(x + e);
  v4f o = xv * g;
  __builtin_nontemporal_store(o, (v4f*)(out + e));
}

// ---------------------------------------------------------------------------
// Host orchestration
// ---------------------------------------------------------------------------
static inline size_t alup(size_t x) { return (x + 255) & ~(size_t)255; }

extern "C" void kernel_launch(void* const* d_in, const int* in_sizes, int n_in,
                              void* d_out, int out_size, void* d_ws, size_t ws_size,
                              hipStream_t stream) {
  const float* x   = (const float*)d_in[0];
  const float* w1  = (const float*)d_in[1];
  const float* b1  = (const float*)d_in[2];
  const float* w2  = (const float*)d_in[3];
  const float* b2  = (const float*)d_in[4];
  const float* w3  = (const float*)d_in[5];
  const float* b3  = (const float*)d_in[6];
  const float* wc  = (const float*)d_in[7];
  const float* bc  = (const float*)d_in[8];
  const float* gam = (const float*)d_in[9];
  const float* bet = (const float*)d_in[10];
  const float* mu  = (const float*)d_in[11];
  const float* var = (const float*)d_in[12];
  const float* cb  = (const float*)d_in[13];
  const float* sm  = (const float*)d_in[14];
  const float* fcw = (const float*)d_in[15];
  const float* fcb = (const float*)d_in[16];
  float* out = (float*)d_out;
  char* ws = (char*)d_ws;

  // workspace layout
  size_t o = 0;
  size_t o_xh = o; o += alup((size_t)BB * CC * NPIX * 2);
  size_t o_e1 = o; o += alup((size_t)BB * CRr * NPIX * 2);
  size_t o_e2 = o; o += alup((size_t)BB * CRr * NPIX * 2);
  size_t o_e3 = o; o += alup((size_t)BB * CC * NPIX * 2);
  size_t o_xc = o; o += alup((size_t)BB * CC * NPIX * 2);
  size_t o_w1 = o; o += alup((size_t)CRr * CC * 2);
  size_t o_w2 = o; o += alup((size_t)9 * CRr * CRr * 2);
  size_t o_w3 = o; o += alup((size_t)CC * CRr * 2);
  size_t o_wc = o; o += alup((size_t)9 * CC * CC * 2);
  size_t o_wt = o; o += alup((size_t)BB * NPIX * KCB * 4);
  size_t o_xn = o; o += alup((size_t)BB * NPIX * 4);
  size_t o_cn = o; o += alup((size_t)KCB * 4);
  size_t o_ek = o; o += alup((size_t)BB * KCB * CC * 4);
  size_t o_ev = o; o += alup((size_t)BB * CC * 4);
  (void)ws_size; (void)n_in; (void)in_sizes; (void)out_size;

  _Float16* xh  = (_Float16*)(ws + o_xh);
  _Float16* e1h = (_Float16*)(ws + o_e1);
  _Float16* e2h = (_Float16*)(ws + o_e2);
  _Float16* e3h = (_Float16*)(ws + o_e3);
  _Float16* xch = (_Float16*)(ws + o_xc);
  _Float16* w1h = (_Float16*)(ws + o_w1);
  _Float16* w2p = (_Float16*)(ws + o_w2);
  _Float16* w3h = (_Float16*)(ws + o_w3);
  _Float16* wcp = (_Float16*)(ws + o_wc);
  float* wtb = (float*)(ws + o_wt);
  float* xnb = (float*)(ws + o_xn);
  float* cnb = (float*)(ws + o_cn);
  float* ekb = (float*)(ws + o_ek);
  float* evb = (float*)(ws + o_ev);

  // 1) cast x + weights to f16 (pack 3x3 weights per-tap)
  {
    int n = BB * CC * NPIX;
    cvt_f32_f16_kernel<<<dim3((n + 255) / 256), dim3(256), 0, stream>>>(x, xh, n);
  }
  cvt_f32_f16_kernel<<<dim3(32), dim3(256), 0, stream>>>(w1, w1h, CRr * CC);
  pack_w3x3_kernel<<<dim3(36), dim3(256), 0, stream>>>(w2, w2p, CRr, CRr);
  cvt_f32_f16_kernel<<<dim3(32), dim3(256), 0, stream>>>(w3, w3h, CC * CRr);
  pack_w3x3_kernel<<<dim3(2304), dim3(256), 0, stream>>>(wc, wcp, CC, CC);

  // 2) encoder
  conv1x1_wmma_kernel<<<dim3(32, CRr / 16, BB), dim3(256), 0, stream>>>(
      xh, w1h, b1, e1h, CC, CRr);
  conv3x3_wmma_kernel<<<dim3(32, CRr / 16, BB), dim3(256), 0, stream>>>(
      e1h, w2p, b2, gam, bet, mu, var, e2h, CRr, CRr, 0);
  conv1x1_wmma_kernel<<<dim3(32, CC / 16, BB), dim3(256), 0, stream>>>(
      e2h, w3h, b3, e3h, CRr, CC);

  // 3) CBR block with fused BN+ReLU (the 77-GFLOP conv)
  conv3x3_wmma_kernel<<<dim3(32, CC / 16, BB), dim3(256), 0, stream>>>(
      e3h, wcp, bc, gam, bet, mu, var, xch, CC, CC, 1);

  // 4) codebook soft-assignment
  xnorm_kernel<<<dim3(NPIX / 256, BB), dim3(256), 0, stream>>>(xch, xnb);
  cnorm_kernel<<<dim3(1), dim3(32), 0, stream>>>(cb, cnb);
  dist_softmax_kernel<<<dim3(32, 1, BB), dim3(256), 0, stream>>>(
      xch, cb, xnb, cnb, sm, wtb);
  ek_wmma_kernel<<<dim3(2, 1, BB), dim3(256), 0, stream>>>(wtb, xch, ekb);

  // 5) FC + sigmoid gate, then elementwise output
  fc_sigmoid_kernel<<<dim3(BB), dim3(CC), 0, stream>>>(ekb, fcw, fcb, evb);
  {
    size_t total4 = (size_t)BB * CC * NPIX / 4;
    gate_kernel<<<dim3((unsigned)(total4 / 256)), dim3(256), 0, stream>>>(
        x, evb, out);
  }
}
